// Embedding2Score_46239617909196
// MI455X (gfx1250) — matrix operand order repaired and checked by reading, not compiled, not run on hardware
//
#include <hip/hip_runtime.h>
#include <hip/hip_bf16.h>
#include <cstdint>

#define H 128

typedef __attribute__((ext_vector_type(16))) __bf16 v16bf;
typedef __attribute__((ext_vector_type(8)))  __bf16 v8bf;
typedef __attribute__((ext_vector_type(8)))  float  v8f;

static __device__ __forceinline__ v16bf cat8(v8bf a, v8bf b) {
  return __builtin_shufflevector(a, b, 0,1,2,3,4,5,6,7,8,9,10,11,12,13,14,15);
}

// ---------------- K1: last node index per graph (batch sorted, graphs non-empty)
__global__ void k_last_idx(const int* __restrict__ batch, int* __restrict__ last_idx, int N) {
  int i = blockIdx.x * blockDim.x + threadIdx.x;
  if (i >= N) return;
  int b = batch[i];
  if (i == N - 1 || batch[i + 1] != b) last_idx[b] = i;
}

// ---------------- K2: h1[g] = v_n[g] @ W1 + W1_b + W2_b  (biases folded)
__global__ void k_h1(const float* __restrict__ emb, const float* __restrict__ W1,
                     const float* __restrict__ b1, const float* __restrict__ b2,
                     const int* __restrict__ last_idx, float* __restrict__ h1) {
  __shared__ float v[H];
  int g = blockIdx.x, j = threadIdx.x;
  int li = last_idx[g];
  v[j] = emb[(size_t)li * H + j];
  __syncthreads();
  float acc = b1[j] + b2[j];
#pragma unroll 8
  for (int k = 0; k < H; ++k) acc = fmaf(v[k], W1[k * H + j], acc);
  h1[(size_t)g * H + j] = acc;
}

// ---------------- K3: coef[i] = num_count[i] * (q . sigmoid(h1[batch[i]] + x_i @ W2) + q_b)
// W2 rows 0..126 live in LDS; row 127 lives in a per-thread register so
// (127*128 + 128) floats == exactly 64KB dynamic LDS.
__global__ void k_coef(const float* __restrict__ emb, const float* __restrict__ W2,
                       const float* __restrict__ h1, const float* __restrict__ qw,
                       const float* __restrict__ qb, const float* __restrict__ nc,
                       const int* __restrict__ batch, float* __restrict__ coef, int N) {
  extern __shared__ float smem[];
  float* W2s = smem;            // 127*128 floats
  float* red = smem + 127 * H;  // 128 floats
  int t = threadIdx.x;
  for (int idx = t; idx < 127 * H; idx += H) W2s[idx] = W2[idx];
  float w2last = W2[127 * H + t];
  float qwt = qw[t];
  float qbv = qb[0];
  __syncthreads();
  for (int i = blockIdx.x; i < N; i += gridDim.x) {
    const float* x = emb + (size_t)i * H;   // uniform address -> scalar loads
    int g = batch[i];
    float acc = h1[(size_t)g * H + t];
#pragma unroll 4
    for (int k = 0; k < 127; ++k) acc = fmaf(x[k], W2s[k * H + t], acc);
    acc = fmaf(x[127], w2last, acc);
    float gate = 1.f / (1.f + expf(-acc));
    red[t] = gate * qwt;
    __syncthreads();
    for (int s = 64; s > 0; s >>= 1) {
      if (t < s) red[t] += red[t + s];
      __syncthreads();
    }
    if (t == 0) coef[i] = nc[i] * (red[0] + qbv);
    __syncthreads();
  }
}

// ---------------- K4: s_g (segmented sum), s_h = [v_n, s_g] @ W3 + b3, split to bf16 hi/lo
__global__ void k_sh(const float* __restrict__ emb, const float* __restrict__ coef,
                     const int* __restrict__ last_idx,
                     const float* __restrict__ W3, const float* __restrict__ b3,
                     __bf16* __restrict__ A_hi, __bf16* __restrict__ A_lo) {
  __shared__ float vn[H];
  __shared__ float sg[H];
  int g = blockIdx.x, t = threadIdx.x;
  int end = last_idx[g];
  int start = (g == 0) ? 0 : (last_idx[g - 1] + 1);
  float s = 0.f;
  for (int i = start; i <= end; ++i) s = fmaf(coef[i], emb[(size_t)i * H + t], s);
  sg[t] = s;
  vn[t] = emb[(size_t)end * H + t];
  __syncthreads();
  float acc = b3[t];
#pragma unroll 4
  for (int k = 0; k < H; ++k) acc = fmaf(vn[k], W3[k * H + t], acc);
#pragma unroll 4
  for (int k = 0; k < H; ++k) acc = fmaf(sg[k], W3[(H + k) * H + t], acc);
  __bf16 hb = (__bf16)acc;
  A_hi[(size_t)g * H + t] = hb;
  A_lo[(size_t)g * H + t] = (__bf16)(acc - (float)hb);
}

// ---------------- K5: out[G,V] = s_h @ item_weight^T via split-bf16 WMMA (f32-accurate)
// Block = 256 threads = 8 waves; each wave owns one 16-row M tile (128 rows/block).
// B tile (16 item rows x 128 k) converted once per N-tile into LDS, shared by 8 waves.
__global__ __launch_bounds__(256) void k_gemm(
    const __bf16* __restrict__ A_hi, const __bf16* __restrict__ A_lo,
    const float* __restrict__ Bmat, float* __restrict__ out,
    int G, int V, int nTiles) {
  __shared__ __bf16 bhi_s[16][136];   // +8 halves pad to spread LDS banks
  __shared__ __bf16 blo_s[16][136];
  const int t = threadIdx.x;
  const int lane = t & 31;
  const int wave = t >> 5;
  const int sel = lane >> 4;        // which K half-chunk this lane holds
  const int l16 = lane & 15;
  const int m_base = (blockIdx.x * 8 + wave) * 16;
  const bool active = (m_base + 16 <= G);   // wave-uniform -> EXEC all-ones inside

  // A fragments: ISA 16-bit A 16x32 layout. Lane L holds row M=L%16,
  // halves K = c*32 + sel*8 + {0..7} then c*32 + 16 + sel*8 + {0..7}.
  v16bf a_hi[4], a_lo[4];
  if (active) {
    const int row = m_base + l16;
#pragma unroll
    for (int c = 0; c < 4; ++c) {
      int k0 = c * 32 + sel * 8;
      a_hi[c] = cat8(*(const v8bf*)(A_hi + (size_t)row * H + k0),
                     *(const v8bf*)(A_hi + (size_t)row * H + k0 + 16));
      a_lo[c] = cat8(*(const v8bf*)(A_lo + (size_t)row * H + k0),
                     *(const v8bf*)(A_lo + (size_t)row * H + k0 + 16));
    }
  }

  const int crow = t >> 4;        // 0..15 : item row within tile
  const int ck0  = (t & 15) * 8;  // 0..120: 8 k-values per thread

  for (int vt = blockIdx.y; vt < nTiles; vt += gridDim.y) {
    // stage B tile as hi/lo bf16 in LDS (fp32 -> split once, reused by 8 waves)
    {
      const float* src = Bmat + (size_t)(vt * 16 + crow) * H + ck0;
      float4 f0 = *(const float4*)(src);
      float4 f1 = *(const float4*)(src + 4);
      float fv[8] = {f0.x, f0.y, f0.z, f0.w, f1.x, f1.y, f1.z, f1.w};
#pragma unroll
      for (int e = 0; e < 8; ++e) {
        float f = fv[e];
        __bf16 hb = (__bf16)f;
        bhi_s[crow][ck0 + e] = hb;
        blo_s[crow][ck0 + e] = (__bf16)(f - (float)hb);
      }
    }
    __syncthreads();

    if (active) {
      v8f acc = {};
#pragma unroll
      for (int c = 0; c < 4; ++c) {
        // B 32x16 layout: lane n holds N=n, halves K = c*32 + sel*8 + {0..7} then c*32 + 16 + sel*8 + {0..7}
        int off = c * 32 + sel * 8;
        v16bf b_hi = cat8(*(const v8bf*)&bhi_s[l16][off],
                          *(const v8bf*)&bhi_s[l16][off + 16]);
        v16bf b_lo = cat8(*(const v8bf*)&blo_s[l16][off],
                          *(const v8bf*)&blo_s[l16][off + 16]);
        // split product: lo*hi + hi*lo + hi*hi, f32 accumulate (~fp32 accuracy)
        acc = __builtin_amdgcn_wmma_f32_16x16x32_bf16(false, a_lo[c], false, b_hi,
                                                      (short)0, acc, false, false);
        acc = __builtin_amdgcn_wmma_f32_16x16x32_bf16(false, a_hi[c], false, b_lo,
                                                      (short)0, acc, false, false);
        acc = __builtin_amdgcn_wmma_f32_16x16x32_bf16(false, a_hi[c], false, b_hi,
                                                      (short)0, acc, false, false);
      }
      // C/D layout: VGPR r -> (M = m_base + r + 8*sel, N = l16). NT stores: output
      // is write-once, keep it out of L2 so item_weight stays resident.
      const size_t col = (size_t)vt * 16 + l16;
#pragma unroll
      for (int r = 0; r < 8; ++r) {
        int m = m_base + r + sel * 8;
        __builtin_nontemporal_store(acc[r], out + (size_t)m * V + col);
      }
    }
    __syncthreads();   // before next iteration overwrites the LDS tile
  }
}

// ---------------- host launcher
extern "C" void kernel_launch(void* const* d_in, const int* in_sizes, int n_in,
                              void* d_out, int out_size, void* d_ws, size_t ws_size,
                              hipStream_t stream) {
  const float* emb   = (const float*)d_in[0];
  const float* item  = (const float*)d_in[1];
  const float* W1    = (const float*)d_in[2];
  const float* b1    = (const float*)d_in[3];
  const float* W2    = (const float*)d_in[4];
  const float* b2    = (const float*)d_in[5];
  const float* qw    = (const float*)d_in[6];
  const float* qb    = (const float*)d_in[7];
  const float* W3    = (const float*)d_in[8];
  const float* b3    = (const float*)d_in[9];
  const float* nc    = (const float*)d_in[10];
  const int*   batch = (const int*)d_in[11];
  (void)n_in; (void)ws_size;

  const int N = in_sizes[10];
  const int V = in_sizes[1] / H;
  const int G = (V > 0) ? (out_size / V) : 0;
  float* out = (float*)d_out;

  uintptr_t p = (uintptr_t)d_ws;
  auto carve = [&](size_t bytes) -> void* {
    uintptr_t q = p;
    p = (p + bytes + 255) & ~(uintptr_t)255;
    return (void*)q;
  };
  int*    last_idx = (int*)carve((size_t)G * sizeof(int));
  float*  h1       = (float*)carve((size_t)G * H * sizeof(float));
  float*  coef     = (float*)carve((size_t)N * sizeof(float));
  __bf16* A_hi     = (__bf16*)carve((size_t)G * H * sizeof(__bf16));
  __bf16* A_lo     = (__bf16*)carve((size_t)G * H * sizeof(__bf16));

  k_last_idx<<<(N + 255) / 256, 256, 0, stream>>>(batch, last_idx, N);
  k_h1<<<G, H, 0, stream>>>(emb, W1, b1, b2, last_idx, h1);
  int coefBlocks = (N < 2048) ? (N > 0 ? N : 1) : 2048;
  k_coef<<<coefBlocks, H, 65536, stream>>>(emb, W2, h1, qw, qb, nc, batch, coef, N);
  k_sh<<<G, H, 0, stream>>>(emb, coef, last_idx, W3, b3, A_hi, A_lo);

  int nTiles = V / 16;                       // V = 100000 -> 6250 exact tiles
  dim3 grid((G + 127) / 128, 625);           // 16 M-blocks x 625 N-strips
  k_gemm<<<grid, 256, 0, stream>>>(A_hi, A_lo, item, out, G, V, nTiles);
}
